// MultiConv_43447889166896
// MI455X (gfx1250) — matrix operand order repaired and checked by reading, not compiled, not run on hardware
//
#include <hip/hip_runtime.h>
#include <hip/hip_bf16.h>

// Problem constants (match reference)
#define BATCH 2
#define NPB   131072
#define NTOT  (BATCH * NPB)      // 262144
#define CIN   96
#define COUT  96
#define K3    27
#define HID   48                 // COUT / RED
#define EPSV  1e-5f

typedef __attribute__((ext_vector_type(16))) __bf16 bf16x16;
typedef __attribute__((ext_vector_type(8)))  float  f32x8;

union FragU {
    bf16x16 v;
    uint4   u[2];
    unsigned int w[8];
};

__device__ __forceinline__ unsigned int pk_bf(float a, float b) {
    unsigned int ua = __float_as_uint(a);
    ua = (ua + 0x7FFFu + ((ua >> 16) & 1u)) >> 16;     // RNE
    unsigned int ub = __float_as_uint(b);
    ub = (ub + 0x7FFFu + ((ub >> 16) & 1u)) >> 16;
    return ua | (ub << 16);
}

// ---------------------------------------------------------------- K0a: stats init + zero sentinel row
__global__ void init_stats(float* gsum, float* gsq, int* gmax, unsigned int* xzrow) {
    int t = blockIdx.x * blockDim.x + threadIdx.x;
    if (t < BATCH * COUT) {
        gsum[t] = 0.f;
        gsq[t]  = 0.f;
        gmax[t] = 0xFF800000;    // -inf bits
    }
    if (t < CIN / 2) xzrow[t] = 0u;   // sentinel zero row (row NTOT of xbf)
}

// ---------------------------------------------------------------- K0b: x -> bf16
__global__ void cvt_x(const float* __restrict__ x, unsigned int* __restrict__ xbf2) {
    long i = (long)blockIdx.x * blockDim.x + threadIdx.x;     // over NTOT*24 float4s
    if (i >= (long)NTOT * (CIN / 4)) return;
    float4 f = ((const float4*)x)[i];
    uint2 o;
    o.x = pk_bf(f.x, f.y);
    o.y = pk_bf(f.z, f.w);
    ((uint2*)xbf2)[i] = o;
}

// ---------------------------------------------------------------- K0c: W -> bf16 B-fragments
// Layout: wfrag[((k*3+kc)*6 + t) * 256 + lane*8 + v]
// Fragment element (lane, v): N = lane%16 -> cout = 16*t + N
//                             K = 16*(lane/16) + 2v (+1 in high half) -> cin = 32*kc + K
__global__ void pack_w(const float* __restrict__ W, unsigned int* __restrict__ wfrag) {
    int id = blockIdx.x * blockDim.x + threadIdx.x;
    if (id >= K3 * 3 * 6 * 32) return;
    int lane = id & 31;
    int rest = id >> 5;
    int t  = rest % 6;  rest /= 6;
    int kc = rest % 3;
    int k  = rest / 3;
    int half = lane >> 4;
    int cout = 16 * t + (lane & 15);
    #pragma unroll
    for (int v = 0; v < 8; ++v) {
        int kk = 32 * kc + 16 * half + 2 * v;
        float w0 = W[((long)k * CIN + kk)     * COUT + cout];
        float w1 = W[((long)k * CIN + kk + 1) * COUT + cout];
        wfrag[(long)id * 8 + v] = pk_bf(w0, w1);
    }
}

// ---------------------------------------------------------------- K1: sparse conv via WMMA
// One wave (32 lanes) computes a 32-point x 96-cout block with 12 f32 accumulators.
// Invalid neighbors are redirected to the all-zero sentinel row xbf[NTOT].
__global__ __launch_bounds__(256) void conv_wmma(
    const unsigned short* __restrict__ xbf,
    const int*            __restrict__ nbr,
    const unsigned int*   __restrict__ wfrag,
    float*                __restrict__ hout,
    float* __restrict__ gsum, float* __restrict__ gsq, int* __restrict__ gmax)
{
    __shared__ float smSum[COUT];
    __shared__ float smSq[COUT];
    __shared__ int   smMax[COUT];

    int tid = threadIdx.x;
    if (tid < COUT) { smSum[tid] = 0.f; smSq[tid] = 0.f; smMax[tid] = 0xFF800000; }
    __syncthreads();

    int wave = tid >> 5;
    int lane = tid & 31;
    int half = lane >> 4;
    int r    = lane & 15;
    long rowbase = ((long)blockIdx.x * 8 + wave) * 32;

    f32x8 acc[12];
    #pragma unroll
    for (int i = 0; i < 12; ++i) {
        f32x8 z = {0.f, 0.f, 0.f, 0.f, 0.f, 0.f, 0.f, 0.f};
        acc[i] = z;
    }

    for (int k = 0; k < K3; ++k) {
        int idx0 = nbr[(rowbase + r)      * K3 + k];
        int idx1 = nbr[(rowbase + 16 + r) * K3 + k];
        idx0 = (idx0 < 0) ? NTOT : idx0;               // sentinel zero row
        idx1 = (idx1 < 0) ? NTOT : idx1;
        const uint4* row0 = (const uint4*)(xbf + (long)idx0 * CIN);
        const uint4* row1 = (const uint4*)(xbf + (long)idx1 * CIN);

        #pragma unroll
        for (int kc = 0; kc < 3; ++kc) {
            FragU a0, a1;
            // A frag: 16 contiguous bf16 at channel 32*kc + 16*half (two b128 loads)
            a0.u[0] = row0[4 * kc + 2 * half];
            a0.u[1] = row0[4 * kc + 2 * half + 1];
            a1.u[0] = row1[4 * kc + 2 * half];
            a1.u[1] = row1[4 * kc + 2 * half + 1];

            const uint4* bp = (const uint4*)(wfrag + (((long)(k * 3 + kc) * 6) << 8)) + (lane << 1);
            #pragma unroll
            for (int t = 0; t < 6; ++t) {
                FragU bb;
                bb.u[0] = bp[0];
                bb.u[1] = bp[1];
                bp += 64;   // next cout tile: 256 dwords
                acc[t] = __builtin_amdgcn_wmma_f32_16x16x32_bf16(
                    false, a0.v, false, bb.v, (short)0, acc[t], false, false);
                acc[6 + t] = __builtin_amdgcn_wmma_f32_16x16x32_bf16(
                    false, a1.v, false, bb.v, (short)0, acc[6 + t], false, false);
            }
        }
    }

    // Store h (conv output) + per-channel partial stats.
    #pragma unroll
    for (int t = 0; t < 6; ++t) {
        int c = 16 * t + r;
        float s = 0.f, q = 0.f, m = -3.4e38f;
        #pragma unroll
        for (int v = 0; v < 8; ++v) {
            long row = rowbase + v + 8 * half;          // D layout: M = v + 8*half, N = lane%16
            float e0 = acc[t][v];
            float e1 = acc[6 + t][v];
            hout[row * COUT + c]        = e0;
            hout[(row + 16) * COUT + c] = e1;
            s += e0 + e1;
            q += e0 * e0 + e1 * e1;
            m = fmaxf(m, fmaxf(e0, e1));
        }
        s += __shfl_xor(s, 16, 32);
        q += __shfl_xor(q, 16, 32);
        m = fmaxf(m, __shfl_xor(m, 16, 32));
        if (lane < 16) {
            atomicAdd(&smSum[c], s);
            atomicAdd(&smSq[c], q);
            if (m >= 0.f) atomicMax(&smMax[c], __float_as_int(m));
            else          atomicMin((unsigned int*)&smMax[c], __float_as_uint(m));
        }
    }
    __syncthreads();

    if (tid < COUT) {
        int b = ((long)blockIdx.x * 256 >= (long)NPB) ? 1 : 0;   // whole block in one batch
        atomicAdd(&gsum[b * COUT + tid], smSum[tid]);
        atomicAdd(&gsq[b * COUT + tid], smSq[tid]);
        float m = __int_as_float(smMax[tid]);
        if (m >= 0.f) atomicMax(&gmax[b * COUT + tid], __float_as_int(m));
        else          atomicMin((unsigned int*)&gmax[b * COUT + tid], __float_as_uint(m));
    }
}

// ---------------------------------------------------------------- K2: attention MLPs + analytic BN stats
__global__ void attn_stats(
    const float* __restrict__ gsum, const float* __restrict__ gsq, const int* __restrict__ gmax,
    const float* __restrict__ w1, const float* __restrict__ b1,
    const float* __restrict__ w2, const float* __restrict__ b2,
    const float* __restrict__ gamma, const float* __restrict__ beta,
    float* __restrict__ coef, float* __restrict__ bshift)
{
    __shared__ float vec[4][COUT];   // mean0, max0, mean1, max1
    __shared__ float hid[4][HID];
    __shared__ float z[2][COUT];
    __shared__ float smx[2], ssum[2];
    int t = threadIdx.x;

    if (t < COUT) {
        vec[0][t] = gsum[t] * (1.f / NPB);
        vec[1][t] = __int_as_float(gmax[t]);
        vec[2][t] = gsum[COUT + t] * (1.f / NPB);
        vec[3][t] = __int_as_float(gmax[COUT + t]);
    }
    __syncthreads();

    if (t < HID) {
        #pragma unroll
        for (int v = 0; v < 4; ++v) {
            float s = b1[t];
            for (int c = 0; c < COUT; ++c) s += vec[v][c] * w1[c * HID + t];
            hid[v][t] = fmaxf(s, 0.f);
        }
    }
    __syncthreads();

    if (t < COUT) {
        #pragma unroll
        for (int b = 0; b < 2; ++b) {
            float zz = 0.f;
            #pragma unroll
            for (int u = 0; u < 2; ++u) {
                float s = b2[t];
                for (int j = 0; j < HID; ++j) s += hid[2 * b + u][j] * w2[j * COUT + t];
                zz += fmaxf(s, 0.f);
            }
            z[b][t] = zz;
        }
    }
    __syncthreads();

    if (t < 2) {
        float mx = -3.4e38f;
        for (int c = 0; c < COUT; ++c) mx = fmaxf(mx, z[t][c]);
        float s = 0.f;
        for (int c = 0; c < COUT; ++c) s += __expf(z[t][c] - mx);
        smx[t] = mx; ssum[t] = s;
    }
    __syncthreads();

    if (t < COUT) {
        float a0 = __expf(z[0][t] - smx[0]) / ssum[0];
        float a1 = __expf(z[1][t] - smx[1]) / ssum[1];
        // out[n,c] = h[n,c] * attn[b,c]  ->  BN stats from per-(b,c) sums of h
        float mu  = (a0 * gsum[t] + a1 * gsum[COUT + t]) * (1.f / (float)NTOT);
        float ex2 = (a0 * a0 * gsq[t] + a1 * a1 * gsq[COUT + t]) * (1.f / (float)NTOT);
        float var = ex2 - mu * mu;
        float A = gamma[t] * rsqrtf(var + EPSV);
        float B = beta[t] - mu * A;
        coef[t]        = a0 * A;
        coef[COUT + t] = a1 * A;
        bshift[t]      = B;
    }
}

// ---------------------------------------------------------------- K3: finalize (in-place on d_out)
__global__ void finalize(const float* __restrict__ x,
                         const float* __restrict__ coef, const float* __restrict__ bshift,
                         float* __restrict__ out)
{
    long i = (long)blockIdx.x * blockDim.x + threadIdx.x;     // over NTOT*24 float4s
    if (i >= (long)NTOT * (COUT / 4)) return;
    long rowq = i / (COUT / 4);
    int  c    = (int)(i % (COUT / 4)) * 4;
    int  b    = (rowq >= (long)NPB) ? 1 : 0;
    float4 h4 = ((const float4*)out)[i];
    float4 x4 = ((const float4*)x)[i];
    const float* cf = coef + b * COUT;
    float4 r;
    r.x = fmaxf(fmaxf(cf[c + 0] * h4.x + bshift[c + 0], 0.f) + x4.x, 0.f);
    r.y = fmaxf(fmaxf(cf[c + 1] * h4.y + bshift[c + 1], 0.f) + x4.y, 0.f);
    r.z = fmaxf(fmaxf(cf[c + 2] * h4.z + bshift[c + 2], 0.f) + x4.z, 0.f);
    r.w = fmaxf(fmaxf(cf[c + 3] * h4.w + bshift[c + 3], 0.f) + x4.w, 0.f);
    ((float4*)out)[i] = r;
}

// ---------------------------------------------------------------- host launch
extern "C" void kernel_launch(void* const* d_in, const int* in_sizes, int n_in,
                              void* d_out, int out_size, void* d_ws, size_t ws_size,
                              hipStream_t stream) {
    const float* x     = (const float*)d_in[0];
    const int*   nbr   = (const int*)  d_in[1];
    const float* Wc    = (const float*)d_in[2];
    const float* w1    = (const float*)d_in[3];
    const float* b1    = (const float*)d_in[4];
    const float* w2    = (const float*)d_in[5];
    const float* b2    = (const float*)d_in[6];
    const float* gamma = (const float*)d_in[7];
    const float* beta  = (const float*)d_in[8];
    float* out = (float*)d_out;

    char*  wsB    = (char*)d_ws;
    float* wsF    = (float*)d_ws;
    float* gsum   = wsF;                   // 192 f
    float* gsq    = wsF + 192;             // 192 f
    int*   gmax   = (int*)(wsF + 384);     // 192 i
    float* coef   = wsF + 576;             // 192 f
    float* bshift = wsF + 768;             // 96 f
    unsigned int*   wfrag = (unsigned int*)  (wsB + 4096);         // 497,664 B
    unsigned short* xbf   = (unsigned short*)(wsB + 512 * 1024);   // (NTOT+1)*96*2 B

    // sentinel zero row lives at xbf + NTOT*CIN
    unsigned int* xzrow = (unsigned int*)(xbf + (long)NTOT * CIN);

    init_stats<<<1, 256, 0, stream>>>(gsum, gsq, gmax, xzrow);

    long nq = (long)NTOT * (CIN / 4);                          // 6,291,456
    cvt_x<<<(int)((nq + 255) / 256), 256, 0, stream>>>(x, (unsigned int*)xbf);

    int npack = K3 * 3 * 6 * 32;                               // 15,552
    pack_w<<<(npack + 255) / 256, 256, 0, stream>>>(Wc, wfrag);

    conv_wmma<<<NTOT / 256, 256, 0, stream>>>(xbf, nbr, wfrag, out, gsum, gsq, gmax);

    attn_stats<<<1, 128, 0, stream>>>(gsum, gsq, gmax, w1, b1, w2, b2, gamma, beta, coef, bshift);

    finalize<<<(int)((nq + 255) / 256), 256, 0, stream>>>(x, coef, bshift, out);
}
